// GroupedExpertsDeepEP_13864154432369
// MI455X (gfx1250) — matrix operand order repaired and checked by reading, not compiled, not run on hardware
//
#include <hip/hip_runtime.h>
#include <cstdint>
#include <cstddef>

typedef __bf16 bf16;
typedef __attribute__((ext_vector_type(16))) __bf16 v16bf;
typedef __attribute__((ext_vector_type(8)))  __bf16 v8bf;
typedef __attribute__((ext_vector_type(4)))  __bf16 v4bf;
typedef __attribute__((ext_vector_type(8)))  float  v8f;

#define T_TOK   8192
#define DMODEL  2048
#define INTER   1024
#define NEXP    8
#define TOPK    2
#define MT      128
#define MAXROWS 17408   // 136 * 128 >= T*K + E*(MT-1)

#define WAIT_ASYNC4() asm volatile("s_wait_asynccnt 0x4" ::: "memory")
#define WAIT_ASYNC2() asm volatile("s_wait_asynccnt 0x2" ::: "memory")
#define WAIT_ASYNC0() asm volatile("s_wait_asynccnt 0x0" ::: "memory")

static __device__ __forceinline__ v8f wmma_bf16(v16bf a, v16bf b, v8f c) {
    return __builtin_amdgcn_wmma_f32_16x16x32_bf16(false, a, false, b, (short)0, c, false, false);
}

// ---------------- routing / packing ----------------

__global__ void k_init(float* __restrict__ y, int* __restrict__ meta) {
    size_t i = (size_t)blockIdx.x * 256 + threadIdx.x;
    float4 z; z.x = z.y = z.z = z.w = 0.0f;
    ((float4*)y)[i] = z;                       // grid exactly covers T*D/4
    if (blockIdx.x == 0 && threadIdx.x < 32) meta[threadIdx.x] = 0; // counts + cursor
}

__global__ void k_count(const unsigned char* __restrict__ mask,
                        const int* __restrict__ idx, int* __restrict__ counts) {
    int t = blockIdx.x * 256 + threadIdx.x;
    if (t < T_TOK && mask[t]) {
        for (int k = 0; k < TOPK; ++k)
            atomicAdd(&counts[idx[t * TOPK + k]], 1);
    }
}

__global__ void k_scan(const int* __restrict__ counts, int* __restrict__ offs) {
    if (threadIdx.x == 0 && blockIdx.x == 0) {
        int off = 0;
        for (int e = 0; e < NEXP; ++e) {
            offs[e] = off;
            off += (counts[e] + MT - 1) & ~(MT - 1);
        }
        offs[NEXP] = off;
    }
}

__global__ void k_fill(const unsigned char* __restrict__ mask,
                       const int* __restrict__ idx, const float* __restrict__ w,
                       int* __restrict__ cursor, const int* __restrict__ offs,
                       int* __restrict__ rowTok, float* __restrict__ pvals) {
    int t = blockIdx.x * 256 + threadIdx.x;
    if (t < T_TOK && mask[t]) {
        for (int k = 0; k < TOPK; ++k) {
            int e = idx[t * TOPK + k];
            int pos = offs[e] + atomicAdd(&cursor[e], 1);
            rowTok[pos] = t;
            pvals[pos]  = w[t * TOPK + k];
        }
    }
}

__global__ void k_pack(const float* __restrict__ x, const int* __restrict__ offs,
                       const int* __restrict__ counts, const int* __restrict__ rowTok,
                       float* __restrict__ pvals, bf16* __restrict__ A) {
    int r = blockIdx.x;
    int e = 0;
    #pragma unroll
    for (int i = 0; i < NEXP; ++i) if (r >= offs[i + 1]) e = i + 1;
    bool valid = (e < NEXP) && (r < offs[e] + counts[e]);
    bf16* dst = A + (size_t)r * DMODEL;
    if (valid) {
        const float4* src = (const float4*)(x + (size_t)rowTok[r] * DMODEL);
        for (int i = threadIdx.x; i < DMODEL / 4; i += 256) {
            float4 v = src[i];
            v4bf b = { (bf16)v.x, (bf16)v.y, (bf16)v.z, (bf16)v.w };
            *(v4bf*)(dst + i * 4) = b;
        }
    } else {
        v4bf z = { (bf16)0.f, (bf16)0.f, (bf16)0.f, (bf16)0.f };
        for (int i = threadIdx.x; i < DMODEL / 4; i += 256)
            *(v4bf*)(dst + i * 4) = z;
        if (threadIdx.x == 0) pvals[r] = 0.0f;   // padding rows contribute nothing
    }
}

// transpose + fp32->bf16: in = [e][R][C] row-major, out = [e][C][R] (n-major, k-contiguous)
__global__ void k_transpose_cvt(const float* __restrict__ in, bf16* __restrict__ out,
                                int R, int C) {
    __shared__ float tile[32][33];
    const size_t esz = (size_t)R * C;
    const float* ip = in + (size_t)blockIdx.z * esz;
    bf16* op = out + (size_t)blockIdx.z * esz;
    int tx = threadIdx.x, ty = threadIdx.y;
    int c0 = blockIdx.x * 32, r0 = blockIdx.y * 32;
    #pragma unroll
    for (int j = 0; j < 4; ++j)
        tile[ty + 8 * j][tx] = ip[(size_t)(r0 + ty + 8 * j) * C + c0 + tx];
    __syncthreads();
    #pragma unroll
    for (int j = 0; j < 4; ++j)
        op[(size_t)(c0 + ty + 8 * j) * R + r0 + tx] = (bf16)tile[tx][ty + 8 * j];
}

// ---------------- GEMM1: H = A @ Wgup, fused silu(gate)*up*p -> Hidden(bf16) ----------------
// block tile: 128 rows x (128 gate + 128 up) cols; 8 waves (2 M-halves x 4 N-quarters of 32)
// A staged in LDS via async copies, 4-buffer / 3-in-flight pipeline, one barrier per K-step.

__global__ __launch_bounds__(256) void k_gemm1(const bf16* __restrict__ A,
                                               const bf16* __restrict__ Wg,
                                               const float* __restrict__ pvals,
                                               const int* __restrict__ offs,
                                               bf16* __restrict__ Hid) {
    const int m0 = blockIdx.y * MT;
    int e = 0;
    #pragma unroll
    for (int i = 0; i < NEXP; ++i) if (m0 >= offs[i + 1]) e = i + 1;
    if (e >= NEXP) return;
    const int g0 = blockIdx.x * 128;          // gate column base in [0,1024)

    __shared__ bf16  As[4][128][48];          // 4-deep async pipeline, 16B-aligned rows
    __shared__ float ps[128];

    const int tid = threadIdx.x;
    const int wave = tid >> 5, lane = tid & 31;
    const int mq = wave & 1, nq = wave >> 1;
    const int lane16 = lane & 15, laneHi = lane >> 4;

    if (tid < 128) ps[tid] = pvals[m0 + tid];

    const v8f vz = {0.f, 0.f, 0.f, 0.f, 0.f, 0.f, 0.f, 0.f};
    v8f accG[4][2], accU[4][2];
    #pragma unroll
    for (int i = 0; i < 4; ++i) { accG[i][0] = vz; accG[i][1] = vz; accU[i][0] = vz; accU[i][1] = vz; }

    // B fragment bases: lane16 = column n, lanes 16-31 take K+16 (contiguous k order, same as A)
    const bf16* Bg = Wg + ((size_t)e * 2048 + (g0 + nq * 32 + lane16)) * (size_t)DMODEL + laneHi * 16;
    const bf16* Bu = Bg + (size_t)INTER * DMODEL;

    const int sr = tid >> 1, sseg = tid & 1;
    const bf16* Asrc = A + (size_t)(m0 + sr) * DMODEL + sseg * 16;

    auto issue = [&](int b, int k) {
        unsigned d0 = (unsigned)(uintptr_t)&As[b][sr][sseg * 16];  // low 32 bits = LDS offset
        const bf16* g = Asrc + k;
        asm volatile("global_load_async_to_lds_b128 %0, %1, off" :: "v"(d0), "v"(g) : "memory");
        asm volatile("global_load_async_to_lds_b128 %0, %1, off" :: "v"(d0 + 16u), "v"(g + 8) : "memory");
    };

    auto compute = [&](int b, int k0) {
        v16bf bg0 = *(const v16bf*)(Bg + k0);
        v16bf bg1 = *(const v16bf*)(Bg + (size_t)16 * DMODEL + k0);
        v16bf bu0 = *(const v16bf*)(Bu + k0);
        v16bf bu1 = *(const v16bf*)(Bu + (size_t)16 * DMODEL + k0);
        #pragma unroll
        for (int ms = 0; ms < 4; ++ms) {
            union { v16bf v; v8bf h[2]; } a;
            const int row = mq * 64 + ms * 16 + lane16;
            // A layout: lanes0-15 K 0-15 ; lanes16-31 K 16-31 (matches B)
            a.h[0] = *(const v8bf*)&As[b][row][laneHi * 16];
            a.h[1] = *(const v8bf*)&As[b][row][laneHi * 16 + 8];
            accG[ms][0] = wmma_bf16(a.v, bg0, accG[ms][0]);
            accG[ms][1] = wmma_bf16(a.v, bg1, accG[ms][1]);
            accU[ms][0] = wmma_bf16(a.v, bu0, accU[ms][0]);
            accU[ms][1] = wmma_bf16(a.v, bu1, accU[ms][1]);
        }
    };

    constexpr int NS = DMODEL / 32;           // 64 stages
    issue(0, 0); issue(1, 32); issue(2, 64);
    #pragma unroll 4
    for (int s = 0; s < NS - 3; ++s) {        // branchless steady state
        WAIT_ASYNC4();
        __syncthreads();                      // stage s visible WG-wide; buf (s+3)&3 reusable
        issue((s + 3) & 3, (s + 3) * 32);
        compute(s & 3, s * 32);
    }
    WAIT_ASYNC4(); __syncthreads(); compute((NS - 3) & 3, (NS - 3) * 32);
    WAIT_ASYNC2(); __syncthreads(); compute((NS - 2) & 3, (NS - 2) * 32);
    WAIT_ASYNC0(); __syncthreads(); compute((NS - 1) & 3, (NS - 1) * 32);

    #pragma unroll
    for (int ms = 0; ms < 4; ++ms) {
        #pragma unroll
        for (int ns = 0; ns < 2; ++ns) {
            const int icol = g0 + nq * 32 + ns * 16 + lane16;
            #pragma unroll
            for (int r = 0; r < 8; ++r) {
                const int m = mq * 64 + ms * 16 + laneHi * 8 + r;   // lanes16-31 => M+8
                const float p = ps[m];
                const float g = accG[ms][ns][r];
                const float u = accU[ms][ns][r];
                const float s = g / (1.0f + __expf(-g));
                Hid[(size_t)(m0 + m) * INTER + icol] = (bf16)(s * u * p);
            }
        }
    }
}

// ---------------- GEMM2: y[tok] += Hidden @ Wdown ----------------
// block tile: 128 rows x 256 cols; 8 waves (2 M-halves x 4 N-quarters of 64)

__global__ __launch_bounds__(256) void k_gemm2(const bf16* __restrict__ Hid,
                                               const bf16* __restrict__ Wd,
                                               const int* __restrict__ offs,
                                               const int* __restrict__ counts,
                                               const int* __restrict__ rowTok,
                                               float* __restrict__ y) {
    const int m0 = blockIdx.y * MT;
    int e = 0;
    #pragma unroll
    for (int i = 0; i < NEXP; ++i) if (m0 >= offs[i + 1]) e = i + 1;
    if (e >= NEXP) return;
    const int n0 = blockIdx.x * 256;

    __shared__ bf16 As[4][128][48];
    __shared__ int  toks[128];

    const int tid = threadIdx.x;
    const int wave = tid >> 5, lane = tid & 31;
    const int mq = wave & 1, nq = wave >> 1;
    const int lane16 = lane & 15, laneHi = lane >> 4;

    const int base = offs[e], cnt = counts[e];
    if (tid < 128) {
        int gr = m0 + tid;
        toks[tid] = (gr - base < cnt) ? rowTok[gr] : -1;
    }

    const v8f vz = {0.f, 0.f, 0.f, 0.f, 0.f, 0.f, 0.f, 0.f};
    v8f acc[4][4];
    #pragma unroll
    for (int i = 0; i < 4; ++i)
        #pragma unroll
        for (int j = 0; j < 4; ++j) acc[i][j] = vz;

    const bf16* B0 = Wd + ((size_t)e * 2048 + (n0 + nq * 64 + lane16)) * (size_t)INTER + laneHi * 16;

    const int sr = tid >> 1, sseg = tid & 1;
    const bf16* Asrc = Hid + (size_t)(m0 + sr) * INTER + sseg * 16;

    auto issue = [&](int b, int k) {
        unsigned d0 = (unsigned)(uintptr_t)&As[b][sr][sseg * 16];
        const bf16* g = Asrc + k;
        asm volatile("global_load_async_to_lds_b128 %0, %1, off" :: "v"(d0), "v"(g) : "memory");
        asm volatile("global_load_async_to_lds_b128 %0, %1, off" :: "v"(d0 + 16u), "v"(g + 8) : "memory");
    };

    auto compute = [&](int b, int k0) {
        v16bf b0 = *(const v16bf*)(B0 + k0);
        v16bf b1 = *(const v16bf*)(B0 + (size_t)16 * INTER + k0);
        v16bf b2 = *(const v16bf*)(B0 + (size_t)32 * INTER + k0);
        v16bf b3 = *(const v16bf*)(B0 + (size_t)48 * INTER + k0);
        #pragma unroll
        for (int ms = 0; ms < 4; ++ms) {
            union { v16bf v; v8bf h[2]; } a;
            const int row = mq * 64 + ms * 16 + lane16;
            a.h[0] = *(const v8bf*)&As[b][row][laneHi * 16];
            a.h[1] = *(const v8bf*)&As[b][row][laneHi * 16 + 8];
            acc[ms][0] = wmma_bf16(a.v, b0, acc[ms][0]);
            acc[ms][1] = wmma_bf16(a.v, b1, acc[ms][1]);
            acc[ms][2] = wmma_bf16(a.v, b2, acc[ms][2]);
            acc[ms][3] = wmma_bf16(a.v, b3, acc[ms][3]);
        }
    };

    constexpr int NS = INTER / 32;            // 32 stages
    issue(0, 0); issue(1, 32); issue(2, 64);
    #pragma unroll 4
    for (int s = 0; s < NS - 3; ++s) {
        WAIT_ASYNC4();
        __syncthreads();
        issue((s + 3) & 3, (s + 3) * 32);
        compute(s & 3, s * 32);
    }
    WAIT_ASYNC4(); __syncthreads(); compute((NS - 3) & 3, (NS - 3) * 32);
    WAIT_ASYNC2(); __syncthreads(); compute((NS - 2) & 3, (NS - 2) * 32);
    WAIT_ASYNC0(); __syncthreads(); compute((NS - 1) & 3, (NS - 1) * 32);

    #pragma unroll
    for (int ms = 0; ms < 4; ++ms) {
        #pragma unroll
        for (int ns = 0; ns < 4; ++ns) {
            const int col = n0 + nq * 64 + ns * 16 + lane16;
            #pragma unroll
            for (int r = 0; r < 8; ++r) {
                const int m = mq * 64 + ms * 16 + laneHi * 8 + r;
                const int tok = toks[m];
                if (tok >= 0)
                    atomicAdd(&y[(size_t)tok * DMODEL + col], acc[ms][ns][r]);
            }
        }
    }
}

// ---------------- launch ----------------

extern "C" void kernel_launch(void* const* d_in, const int* in_sizes, int n_in,
                              void* d_out, int out_size, void* d_ws, size_t ws_size,
                              hipStream_t stream) {
    const float*         x     = (const float*)d_in[0];
    const unsigned char* mask  = (const unsigned char*)d_in[1];   // jnp bool = 1 byte
    const float*         w     = (const float*)d_in[2];
    const int*           idx   = (const int*)d_in[3];
    const float*         Wgup  = (const float*)d_in[4];
    const float*         Wdown = (const float*)d_in[5];
    float*               y     = (float*)d_out;

    char* ws = (char*)d_ws;
    int*   counts = (int*)(ws + 0);
    int*   cursor = (int*)(ws + 64);
    int*   offs   = (int*)(ws + 128);
    int*   rowTok = (int*)(ws + 256);                       // MAXROWS ints
    float* pvals  = (float*)(ws + 69888);                   // MAXROWS floats
    bf16*  Apack  = (bf16*)(ws + 139776);                   // MAXROWS x 2048 bf16
    bf16*  Hid    = (bf16*)(ws + 71442944);                 // MAXROWS x 1024 bf16
    bf16*  WgT    = (bf16*)(ws + 107094528);                // 8 x 2048 x 2048 bf16 ([e][n][k])
    bf16*  WdT    = (bf16*)(ws + 174203392);                // 8 x 2048 x 1024 bf16 ([e][n][k])

    k_init<<<(T_TOK * DMODEL / 4) / 256, 256, 0, stream>>>(y, counts);
    k_count<<<T_TOK / 256, 256, 0, stream>>>(mask, idx, counts);
    k_scan<<<1, 1, 0, stream>>>(counts, offs);
    k_fill<<<T_TOK / 256, 256, 0, stream>>>(mask, idx, w, cursor, offs, rowTok, pvals);
    k_pack<<<MAXROWS, 256, 0, stream>>>(x, offs, counts, rowTok, pvals, Apack);
    k_transpose_cvt<<<dim3(64, 64, 8), dim3(32, 8), 0, stream>>>(Wgup, WgT, 2048, 2048);
    k_transpose_cvt<<<dim3(64, 32, 8), dim3(32, 8), 0, stream>>>(Wdown, WdT, 1024, 2048);
    k_gemm1<<<dim3(8, MAXROWS / MT), 256, 0, stream>>>(Apack, WgT, pvals, offs, Hid);
    k_gemm2<<<dim3(8, MAXROWS / MT), 256, 0, stream>>>(Hid, WdT, offs, counts, rowTok, y);
}
